// GQA_89653147337095
// MI455X (gfx1250) — matrix operand order repaired and checked by reading, not compiled, not run on hardware
//
#include <hip/hip_runtime.h>
#include <hip/hip_bf16.h>
#include <math.h>

typedef __bf16 v16bf __attribute__((ext_vector_type(16)));
typedef __bf16 v8bf  __attribute__((ext_vector_type(8)));
typedef float  v8f   __attribute__((ext_vector_type(8)));

union ABFrag { v16bf v; v8bf h[2]; };

__device__ __forceinline__ unsigned short f2bf(float f) {
  unsigned u = __float_as_uint(f);
  u += 0x7FFFu + ((u >> 16) & 1u);   // round-to-nearest-even
  return (unsigned short)(u >> 16);
}

__device__ __forceinline__ v8f zero8() {
  v8f z;
  #pragma unroll
  for (int i = 0; i < 8; ++i) z[i] = 0.0f;
  return z;
}

#define BM 128
#define BN 128
#define BK 32
#define S_LEN 2048
#define HD 128

// C[M,N] = A[M,K] (f32) * B[K,N] (f32), bf16 WMMA accumulate f32.
// M,N multiples of 128; K multiple of 32.
__global__ __launch_bounds__(256)
void gemm_f32_bf16wmma(const float* __restrict__ A, const float* __restrict__ B,
                       float* __restrict__ C, int M, int N, int K) {
  __shared__ unsigned short As[BM * BK];   // [m][k]
  __shared__ unsigned short Bs[BN * BK];   // transposed: [n][k]

  const int t    = threadIdx.x;
  const int lane = t & 31;
  const int wave = t >> 5;
  const int hl   = (lane >> 4) & 1;
  const int l15  = lane & 15;
  const int wm   = wave >> 1;   // 0..3 -> 32-row strips
  const int wn   = wave & 1;    // 0..1 -> 64-col strips
  const int m0   = blockIdx.y * BM;
  const int n0   = blockIdx.x * BN;

  v8f acc[2][4];
  #pragma unroll
  for (int i = 0; i < 2; ++i)
    #pragma unroll
    for (int j = 0; j < 4; ++j) acc[i][j] = zero8();

  const int nk = K / BK;
  for (int kt = 0; kt < nk; ++kt) {
    const int k0 = kt * BK;
    __syncthreads();
    // stage A tile: 128x32 f32 -> bf16 LDS
    #pragma unroll
    for (int i = 0; i < 4; ++i) {
      int c   = t + 256 * i;      // 0..1023 chunks of 4
      int row = c >> 3;           // 0..127
      int kc  = (c & 7) * 4;      // 0..28
      float4 f = *(const float4*)(A + (size_t)(m0 + row) * K + k0 + kc);
      ushort4 hh;
      hh.x = f2bf(f.x); hh.y = f2bf(f.y); hh.z = f2bf(f.z); hh.w = f2bf(f.w);
      *(ushort4*)(&As[row * BK + kc]) = hh;
    }
    // stage B tile transposed: read rows of B coalesced, scatter to [n][k]
    #pragma unroll
    for (int i = 0; i < 4; ++i) {
      int c  = t + 256 * i;
      int k  = c >> 5;            // 0..31
      int cj = (c & 31) * 4;      // 0..124
      float4 f = *(const float4*)(B + (size_t)(k0 + k) * N + n0 + cj);
      Bs[(cj + 0) * BK + k] = f2bf(f.x);
      Bs[(cj + 1) * BK + k] = f2bf(f.y);
      Bs[(cj + 2) * BK + k] = f2bf(f.z);
      Bs[(cj + 3) * BK + k] = f2bf(f.w);
    }
    // prefetch next K-step tiles into cache (global_prefetch_b8)
    if (kt + 1 < nk) {
      const int kn = k0 + BK;
      __builtin_prefetch(A + (size_t)(m0 + (t >> 1)) * K + kn + (t & 1) * 16, 0, 1);
      __builtin_prefetch(B + (size_t)(kn + (t >> 3)) * N + n0 + (t & 7) * 16, 0, 1);
    }
    __syncthreads();

    ABFrag a[2], b[4];
    #pragma unroll
    for (int im = 0; im < 2; ++im) {
      int m = wm * 32 + im * 16 + l15;     // A lane row; 16 consecutive K
      a[im].v = *(const v16bf*)(const void*)(&As[m * BK + hl * 16]);
    }
    #pragma unroll
    for (int in = 0; in < 4; ++in) {
      int n = wn * 64 + in * 16 + l15;     // B lane col; 16 consecutive K
      b[in].v = *(const v16bf*)(const void*)(&Bs[n * BK + hl * 16]);
    }
    #pragma unroll
    for (int im = 0; im < 2; ++im)
      #pragma unroll
      for (int in = 0; in < 4; ++in)
        acc[im][in] = __builtin_amdgcn_wmma_f32_16x16x32_bf16(
            false, a[im].v, false, b[in].v, (short)0, acc[im][in], false, false);
  }

  #pragma unroll
  for (int im = 0; im < 2; ++im)
    #pragma unroll
    for (int in = 0; in < 4; ++in) {
      int colg = n0 + wn * 64 + in * 16 + l15;
      #pragma unroll
      for (int r = 0; r < 8; ++r) {
        int rowg = m0 + wm * 32 + im * 16 + r + 8 * hl;
        C[(size_t)rowg * N + colg] = acc[im][in][r];
      }
    }
}

// RoPE on Q (heads 0..31) and K (32..39), passthrough V (40..47);
// split into head-major bf16 buffers.
__global__ __launch_bounds__(256)
void rope_split(const float* __restrict__ qkv, unsigned short* __restrict__ Qb,
                unsigned short* __restrict__ Kb, unsigned short* __restrict__ Vb) {
  size_t tid = (size_t)blockIdx.x * blockDim.x + threadIdx.x;
  int j    = (int)(tid & 63);          // dim pair
  int head = (int)((tid >> 6) % 48);
  int s    = (int)(tid / (48 * 64));
  const float* src = qkv + ((size_t)s * 48 + head) * HD + 2 * j;
  float x0 = src[0], x1 = src[1];
  float y0 = x0, y1 = x1;
  if (head < 40 && j < 32) {
    float inv = __expf(-(float)j * 0.2878231366f);  // ln(1e4)/32
    float ang = (float)s * inv;
    float sn, cs;
    __sincosf(ang, &sn, &cs);
    y0 = x0 * cs - x1 * sn;
    y1 = x1 * cs + x0 * sn;
  }
  unsigned short* dst;
  if (head < 32)      dst = Qb + ((size_t)head * S_LEN + s) * HD + 2 * j;
  else if (head < 40) dst = Kb + ((size_t)(head - 32) * S_LEN + s) * HD + 2 * j;
  else                dst = Vb + ((size_t)(head - 40) * S_LEN + s) * HD + 2 * j;
  dst[0] = f2bf(y0);
  dst[1] = f2bf(y1);
}

// Flash attention (non-causal), GQA: block = (q-block of 128 rows, head).
// 8 waves x 16 query rows; key blocks of 32; WMMA for QK^T and PV.
// K tile staged with CDNA5 async global->LDS DMA (ASYNCcnt).
__global__ __launch_bounds__(256)
void attn_flash(const unsigned short* __restrict__ Qb,
                const unsigned short* __restrict__ Kb,
                const unsigned short* __restrict__ Vb,
                float* __restrict__ Out) {
  __shared__ unsigned short Kt[32 * HD];     // [key][d]
  __shared__ unsigned short VT[HD * 32];     // [d][key] (transposed)
  __shared__ unsigned short Pw[8][16 * 32];  // per-wave P tile [m][key]

  const int t    = threadIdx.x;
  const int lane = t & 31;
  const int wave = t >> 5;
  const int hl   = (lane >> 4) & 1;
  const int l15  = lane & 15;
  const int h    = blockIdx.y;        // Q head 0..31
  const int kvh  = h >> 2;            // GQA group of 4
  const int q0   = blockIdx.x * 128 + wave * 16;
  const float scale = 0.08838834764831845f;  // 1/sqrt(128)

  // Q fragments held in registers for the whole kernel (4 K-steps over d)
  ABFrag qa[4];
  const unsigned short* Qrow = Qb + ((size_t)h * S_LEN + q0 + l15) * HD;
  #pragma unroll
  for (int ks = 0; ks < 4; ++ks) {
    qa[ks].v = *(const v16bf*)(const void*)(Qrow + ks * 32 + hl * 16);
  }

  v8f o[8];
  #pragma unroll
  for (int i = 0; i < 8; ++i) o[i] = zero8();
  float mrow[8], lrow[8];
  #pragma unroll
  for (int r = 0; r < 8; ++r) { mrow[r] = -1e30f; lrow[r] = 0.0f; }

  const unsigned short* Kbase = Kb + (size_t)kvh * S_LEN * HD;
  const unsigned short* Vbase = Vb + (size_t)kvh * S_LEN * HD;

  for (int kb = 0; kb < S_LEN / 32; ++kb) {
    const int key0 = kb * 32;
    __syncthreads();
    // stage K tile [32][128]: async global->LDS DMA, 16B per lane per op
    #pragma unroll
    for (int i = 0; i < 2; ++i) {
      int c   = t + 256 * i;     // 0..511
      int key = c >> 4;          // 0..31
      int dj  = (c & 15) * 8;    // 0..120
      const unsigned short* gp = Kbase + (size_t)(key0 + key) * HD + dj;
      unsigned lds_off = (unsigned)(size_t)(&Kt[key * HD + dj]);
      asm volatile("global_load_async_to_lds_b128 %0, %1, off"
                   :: "v"(lds_off), "v"((unsigned long long)(size_t)gp)
                   : "memory");
    }
    // stage V tile transposed [d][key] (manual: transpose during store)
    #pragma unroll
    for (int i = 0; i < 2; ++i) {
      int c   = t + 256 * i;
      int key = c >> 4;
      int dj  = (c & 15) * 8;
      const unsigned short* vp = Vbase + (size_t)(key0 + key) * HD + dj;
      #pragma unroll
      for (int e = 0; e < 8; ++e) VT[(dj + e) * 32 + key] = vp[e];
    }
    asm volatile("s_wait_asynccnt 0x0" ::: "memory");
    __syncthreads();

    // S = Q K^T : two 16-key n-tiles, 4 K-steps over d
    v8f s0 = zero8(), s1 = zero8();
    #pragma unroll
    for (int ks = 0; ks < 4; ++ks) {
      v16bf b0 = *(const v16bf*)(const void*)(&Kt[(0  + l15) * HD + ks * 32 + hl * 16]);
      v16bf b1 = *(const v16bf*)(const void*)(&Kt[(16 + l15) * HD + ks * 32 + hl * 16]);
      s0 = __builtin_amdgcn_wmma_f32_16x16x32_bf16(false, qa[ks].v, false, b0, (short)0, s0, false, false);
      s1 = __builtin_amdgcn_wmma_f32_16x16x32_bf16(false, qa[ks].v, false, b1, (short)0, s1, false, false);
    }

    // online softmax over this key block; row r+8*hl lives across 16 lanes
    #pragma unroll
    for (int r = 0; r < 8; ++r) {
      float a0 = s0[r] * scale;
      float a1 = s1[r] * scale;
      float v = fmaxf(a0, a1);
      #pragma unroll
      for (int off = 8; off >= 1; off >>= 1) v = fmaxf(v, __shfl_xor(v, off, 32));
      float mn = fmaxf(mrow[r], v);
      float f  = __expf(mrow[r] - mn);
      float p0 = __expf(a0 - mn);
      float p1 = __expf(a1 - mn);
      float rs = p0 + p1;
      #pragma unroll
      for (int off = 8; off >= 1; off >>= 1) rs += __shfl_xor(rs, off, 32);
      lrow[r] = lrow[r] * f + rs;
      mrow[r] = mn;
      #pragma unroll
      for (int tt = 0; tt < 8; ++tt) o[tt][r] *= f;
      int m = r + 8 * hl;
      Pw[wave][m * 32 + l15]      = f2bf(p0);
      Pw[wave][m * 32 + 16 + l15] = f2bf(p1);
    }
    __syncthreads();

    // O += P V : one K-step (32 keys), 8 d-tiles
    ABFrag pa;
    pa.v = *(const v16bf*)(const void*)(&Pw[wave][l15 * 32 + hl * 16]);
    #pragma unroll
    for (int tt = 0; tt < 8; ++tt) {
      v16bf vb = *(const v16bf*)(const void*)(&VT[(tt * 16 + l15) * 32 + hl * 16]);
      o[tt] = __builtin_amdgcn_wmma_f32_16x16x32_bf16(false, pa.v, false, vb, (short)0, o[tt], false, false);
    }
  }

  // normalize and write attn output f32 [S, 32*128]
  #pragma unroll
  for (int tt = 0; tt < 8; ++tt) {
    int col = h * HD + tt * 16 + l15;
    #pragma unroll
    for (int r = 0; r < 8; ++r) {
      int row = q0 + r + 8 * hl;
      Out[(size_t)row * 4096 + col] = o[tt][r] / lrow[r];
    }
  }
}

extern "C" void kernel_launch(void* const* d_in, const int* in_sizes, int n_in,
                              void* d_out, int out_size, void* d_ws, size_t ws_size,
                              hipStream_t stream) {
  (void)in_sizes; (void)n_in; (void)out_size; (void)ws_size;
  const float* x    = (const float*)d_in[0];   // [2048, 4096]
  const float* Wqkv = (const float*)d_in[1];   // [4096, 6144]
  const float* Wout = (const float*)d_in[2];   // [4096, 4096]
  float* out = (float*)d_out;                  // [2048, 4096]

  char* w = (char*)d_ws;
  float*          qkv  = (float*)w;                              // 2048*6144*4 = 50,331,648
  unsigned short* Qb   = (unsigned short*)(w + 50331648);        // 32*2048*128*2 = 16,777,216
  unsigned short* Kb   = (unsigned short*)(w + 67108864);        //  8*2048*128*2 =  4,194,304
  unsigned short* Vb   = (unsigned short*)(w + 71303168);        //  4,194,304
  float*          attn = (float*)(w + 75497472);                 // 2048*4096*4 = 33,554,432

  // 1) qkv = x @ W_qkv
  gemm_f32_bf16wmma<<<dim3(6144 / BN, 2048 / BM), 256, 0, stream>>>(x, Wqkv, qkv, 2048, 6144, 4096);
  // 2) RoPE + split to bf16 head-major Q/K/V
  rope_split<<<(2048 * 48 * 64) / 256, 256, 0, stream>>>(qkv, Qb, Kb, Vb);
  // 3) flash attention
  attn_flash<<<dim3(2048 / 128, 32), 256, 0, stream>>>(Qb, Kb, Vb, attn);
  // 4) out = attn @ W_out
  gemm_f32_bf16wmma<<<dim3(4096 / BN, 2048 / BM), 256, 0, stream>>>(attn, Wout, out, 2048, 4096, 4096);
}
